// Encoder_90555090469272
// MI455X (gfx1250) — matrix operand order repaired and checked by reading, not compiled, not run on hardware
//
#include <hip/hip_runtime.h>
#include <hip/hip_bf16.h>

// ---------------------------------------------------------------------------
// S4 encoder for MI455X (gfx1250, wave32).
//  - Channel-mixing GEMMs via v_wmma_f32_16x16x32_bf16 (f32 accumulate),
//    register-blocked 16x64 per wave (4 accumulators, 4x A-fragment reuse,
//    2.5 b128 loads per WMMA instead of 4).
//  - S4 bidirectional long-conv computed as a diagonal complex linear
//    recurrence (64 states/channel, 2 states per lane of a wave32),
//    avoiding FFTs entirely.
//  - gelu exact (erff), GLU, LayerNorm as elementwise/wave kernels.
// ---------------------------------------------------------------------------

typedef __attribute__((ext_vector_type(16))) __bf16 bf16x16;
typedef __attribute__((ext_vector_type(8)))  __bf16 bf16x8;
typedef __attribute__((ext_vector_type(8)))  float  f32x8;

#define F_IN   12
#define HDIM   128
#define BATCH  8
#define SEQL   4096
#define RDOWN  4
#define NSTATE 64

__device__ __forceinline__ float gelu_exact(float v) {
    return 0.5f * v * (1.0f + erff(v * 0.70710678118654752440f));
}
__device__ __forceinline__ float sigmoidf_(float v) {
    return 1.0f / (1.0f + __expf(-v));
}

// ------------------------------ elementwise --------------------------------
__global__ void k_f32_to_bf16(const float* __restrict__ s, __bf16* __restrict__ d, long n) {
    long i = (long)blockIdx.x * blockDim.x + threadIdx.x;
    if (i < n) d[i] = (__bf16)s[i];
}
__global__ void k_copy(const float* __restrict__ s, float* __restrict__ d, long n) {
    long i = (long)blockIdx.x * blockDim.x + threadIdx.x;
    if (i < n) d[i] = s[i];
}
__global__ void k_add_inplace(float* __restrict__ x, const float* __restrict__ y, long n) {
    long i = (long)blockIdx.x * blockDim.x + threadIdx.x;
    if (i < n) x[i] += y[i];
}
__global__ void k_gelu_inplace(float* __restrict__ x, long n) {
    long i = (long)blockIdx.x * blockDim.x + threadIdx.x;
    if (i < n) x[i] = gelu_exact(x[i]);
}
// X[row,c] += Z[row,c] * sigmoid(Z[row,h+c])
__global__ void k_glu_residual(const float* __restrict__ Z, float* __restrict__ X, long n, int h) {
    long i = (long)blockIdx.x * blockDim.x + threadIdx.x;
    if (i >= n) return;
    long row = i / h; int c = (int)(i % h);
    float a = Z[row * (2L * h) + c];
    float g = Z[row * (2L * h) + h + c];
    X[i] += a * sigmoidf_(g);
}

// ------------------------- BatchNorm + embedding ---------------------------
__global__ void k_bn_embed(const float* __restrict__ x,
                           const float* __restrict__ bw, const float* __restrict__ bb,
                           const float* __restrict__ bm, const float* __restrict__ bv,
                           const float* __restrict__ W, const float* __restrict__ bias,
                           float* __restrict__ out, long n) {
    long i = (long)blockIdx.x * blockDim.x + threadIdx.x;
    if (i >= n) return;
    long row = i / HDIM; int o = (int)(i % HDIM);
    const float* xp = x + row * F_IN;
    float acc = bias[o];
    #pragma unroll
    for (int f = 0; f < F_IN; ++f) {
        float xn = (xp[f] - bm[f]) * rsqrtf(bv[f] + 1e-5f) * bw[f] + bb[f];
        acc += W[o * F_IN + f] * xn;
    }
    out[i] = acc;
}

// ------------------------------ LayerNorm ----------------------------------
// one wave32 per row
__global__ void k_ln_rows(const float* __restrict__ x, const float* __restrict__ g,
                          const float* __restrict__ b, float* __restrict__ y, int h) {
    long row = blockIdx.x;
    int lane = threadIdx.x;
    const float* xr = x + row * (long)h;
    float s = 0.f, s2 = 0.f;
    for (int c = lane; c < h; c += 32) { float v = xr[c]; s += v; s2 += v * v; }
    for (int off = 16; off; off >>= 1) { s += __shfl_xor(s, off); s2 += __shfl_xor(s2, off); }
    float mean = s / (float)h;
    float var  = s2 / (float)h - mean * mean;
    float inv  = rsqrtf(var + 1e-5f);
    float* yr = y + row * (long)h;
    for (int c = lane; c < h; c += 32) yr[c] = (xr[c] - mean) * inv * g[c] + b[c];
}

// --------------------------- S4D recurrence scans --------------------------
// Each wave handles one (batch, channel). 64 complex states => 2 per lane.
// C_eff = 2 * (C_re + i C_im) * (exp(dtA)-1)/A ;  a = exp(dtA)
__device__ __forceinline__ void s4_coeffs(int c, int lane,
        const float* log_dt, const float* A_re_log, const float* A_im,
        const float* Cre, const float* Cim,
        float ar[2], float ai[2], float cr[2], float ci[2]) {
    float dt = __expf(log_dt[c]);
    #pragma unroll
    for (int q = 0; q < 2; ++q) {
        int n = lane + 32 * q;
        float Are = -__expf(A_re_log[c * NSTATE + n]);
        float Aim = A_im[c * NSTATE + n];
        float dr = Are * dt, di = Aim * dt;
        float ea  = __expf(dr);
        float arr = ea * __cosf(di), aii = ea * __sinf(di);
        ar[q] = arr; ai[q] = aii;
        // (a - 1) / A
        float nr = arr - 1.0f, ni = aii;
        float den = Are * Are + Aim * Aim;
        float br = (nr * Are + ni * Aim) / den;
        float bi = (ni * Are - nr * Aim) / den;
        float c_r = Cre[c * NSTATE + n], c_i = Cim[c * NSTATE + n];
        cr[q] = 2.0f * (c_r * br - c_i * bi);
        ci[q] = 2.0f * (c_r * bi + c_i * br);
    }
}

__global__ void k_s4_fwd(const float* __restrict__ U, float* __restrict__ Y,
                         const float* __restrict__ log_dt, const float* __restrict__ A_re_log,
                         const float* __restrict__ A_im,  const float* __restrict__ Cre,
                         const float* __restrict__ Cim, int Lc, int h) {
    int bc = blockIdx.x, b = bc / h, c = bc % h, lane = threadIdx.x;
    float ar[2], ai[2], cr[2], ci[2];
    s4_coeffs(c, lane, log_dt, A_re_log, A_im, Cre, Cim, ar, ai, cr, ci);
    float sr[2] = {0.f, 0.f}, si[2] = {0.f, 0.f};
    const float* up = U + (long)b * Lc * h + c;
    float* yp = Y + (long)b * Lc * h + c;
    for (int l = 0; l < Lc; ++l) {
        float u = up[(long)l * h];
        float acc = 0.f;
        #pragma unroll
        for (int q = 0; q < 2; ++q) {
            float nsr = ar[q] * sr[q] - ai[q] * si[q] + u;
            float nsi = ar[q] * si[q] + ai[q] * sr[q];
            sr[q] = nsr; si[q] = nsi;
            acc += cr[q] * nsr - ci[q] * nsi;
        }
        for (int off = 16; off; off >>= 1) acc += __shfl_xor(acc, off);
        if (lane == 0) yp[(long)l * h] = acc;
    }
}

// backward scan; adds to forward result, adds D*u, applies exact GELU
__global__ void k_s4_bwd(const float* __restrict__ U, float* __restrict__ Y,
                         const float* __restrict__ log_dt, const float* __restrict__ A_re_log,
                         const float* __restrict__ A_im,  const float* __restrict__ Cre,
                         const float* __restrict__ Cim,   const float* __restrict__ Dp,
                         int Lc, int h) {
    int bc = blockIdx.x, b = bc / h, c = bc % h, lane = threadIdx.x;
    float ar[2], ai[2], cr[2], ci[2];
    s4_coeffs(c, lane, log_dt, A_re_log, A_im, Cre, Cim, ar, ai, cr, ci);
    float Dv = Dp[c];
    float tr[2] = {0.f, 0.f}, ti[2] = {0.f, 0.f};
    const float* up = U + (long)b * Lc * h + c;
    float* yp = Y + (long)b * Lc * h + c;
    for (int l = Lc - 1; l >= 0; --l) {
        float u = up[(long)l * h];
        float acc = 0.f;
        #pragma unroll
        for (int q = 0; q < 2; ++q) acc += cr[q] * tr[q] - ci[q] * ti[q];
        for (int off = 16; off; off >>= 1) acc += __shfl_xor(acc, off);
        if (lane == 0) {
            float v = yp[(long)l * h] + acc + Dv * u;
            yp[(long)l * h] = gelu_exact(v);
        }
        #pragma unroll
        for (int q = 0; q < 2; ++q) {                  // t_{l-1} = u[l] + a * t_l
            float ntr = ar[q] * tr[q] - ai[q] * ti[q] + u;
            float nti = ar[q] * ti[q] + ai[q] * tr[q];
            tr[q] = ntr; ti[q] = nti;
        }
    }
}

// ------------------- patch-conv gather (kernel==stride==4) -----------------
// A[(b*Lo+lo)][c*R + r] = x[b][lo*R + r][c]   (K index matches W (O,C,R) flat)
__global__ void k_conv_gather(const float* __restrict__ x, __bf16* __restrict__ A,
                              int Lo, int Cin, long n) {
    long i = (long)blockIdx.x * blockDim.x + threadIdx.x;
    if (i >= n) return;
    int K  = Cin * RDOWN;
    long row = i / K; int k = (int)(i % K);
    int c = k / RDOWN, r = k % RDOWN;
    long b = row / Lo, lo = row % Lo;
    A[i] = (__bf16)x[((long)b * Lo * RDOWN + lo * RDOWN + r) * Cin + c];
}

// ------------------------------- WMMA GEMM ---------------------------------
// C[M,N] = A[M,K](bf16, row-major) * W[N,K]^T(bf16, row-major) + bias[N]
// 256 threads = 8 waves; wave w -> 16(M) x 64(N) register block at
//   (blockIdx.y*128 + 16w, 64*blockIdx.x): 4 accumulators, A fragment reused 4x.
// Operand layouts used here (wave32; contiguous k, same order for A and B):
//  A: lanes 0-15 row M=lr, K = kk+0..15 contiguous; lanes 16-31 K = kk+16..31
//  B: lanes 0-15 col N=lr, K = kk+0..15 contiguous; lanes 16-31 K = kk+16..31
__global__ void __launch_bounds__(256)
k_wmma_gemm_bf16(const __bf16* __restrict__ A, const __bf16* __restrict__ W,
                 const float* __restrict__ bias, float* __restrict__ C,
                 int M, int N, int K) {
    int wave = threadIdx.x >> 5;
    int lane = threadIdx.x & 31;
    int tm = blockIdx.y * 128 + wave * 16;
    int tn = blockIdx.x * 64;
    if (tm >= M) return;                 // wave-uniform: EXEC stays all-ones
    int half = lane >> 4;
    int lr   = lane & 15;
    const __bf16* arow = A + (long)(tm + lr) * K + half * 16;
    const __bf16* wrow0 = W + (long)(tn +  0 + lr) * K + half * 16;
    const __bf16* wrow1 = W + (long)(tn + 16 + lr) * K + half * 16;
    const __bf16* wrow2 = W + (long)(tn + 32 + lr) * K + half * 16;
    const __bf16* wrow3 = W + (long)(tn + 48 + lr) * K + half * 16;
    f32x8 acc0 = {0.f,0.f,0.f,0.f,0.f,0.f,0.f,0.f};
    f32x8 acc1 = acc0, acc2 = acc0, acc3 = acc0;
    for (int kk = 0; kk < K; kk += 32) {
        bf16x8 a0 = *(const bf16x8*)(arow + kk);
        bf16x8 a1 = *(const bf16x8*)(arow + kk + 8);
        bf16x16 av;
        #pragma unroll
        for (int i = 0; i < 8; ++i) { av[i] = a0[i]; av[i + 8] = a1[i]; }
        __builtin_prefetch(arow + kk + 32, 0, 0);   // gfx1250 global_prefetch_b8
        bf16x16 bv0 = *(const bf16x16*)(wrow0 + kk);
        bf16x16 bv1 = *(const bf16x16*)(wrow1 + kk);
        bf16x16 bv2 = *(const bf16x16*)(wrow2 + kk);
        bf16x16 bv3 = *(const bf16x16*)(wrow3 + kk);
        acc0 = __builtin_amdgcn_wmma_f32_16x16x32_bf16(false, av, false, bv0,
                                                       (short)0, acc0, false, false);
        acc1 = __builtin_amdgcn_wmma_f32_16x16x32_bf16(false, av, false, bv1,
                                                       (short)0, acc1, false, false);
        acc2 = __builtin_amdgcn_wmma_f32_16x16x32_bf16(false, av, false, bv2,
                                                       (short)0, acc2, false, false);
        acc3 = __builtin_amdgcn_wmma_f32_16x16x32_bf16(false, av, false, bv3,
                                                       (short)0, acc3, false, false);
    }
    // D layout: VGPR r -> M = r (lanes 0-15) / M = 8+r (lanes 16-31), N = lane&15
    #pragma unroll
    for (int t = 0; t < 4; ++t) {
        f32x8 a = (t == 0) ? acc0 : (t == 1) ? acc1 : (t == 2) ? acc2 : acc3;
        int col = tn + 16 * t + lr;
        float bb = bias[col];
        #pragma unroll
        for (int r = 0; r < 8; ++r) {
            int row = tm + r + 8 * half;
            C[(long)row * N + col] = a[r] + bb;
        }
    }
}

// -------------------------- nearest-exact upsample -------------------------
__global__ void k_upsample(const float* __restrict__ X, float* __restrict__ out, long n) {
    long i = (long)blockIdx.x * blockDim.x + threadIdx.x;
    if (i >= n) return;
    int c = (int)(i % 512);
    long t = i / 512;
    int l = (int)(t % SEQL);
    long b = t / SEQL;
    out[i] = X[((long)b * (SEQL / 16) + l / 16) * 512 + c];
}

// ===========================================================================
extern "C" void kernel_launch(void* const* d_in, const int* in_sizes, int n_in,
                              void* d_out, int out_size, void* d_ws, size_t ws_size,
                              hipStream_t stream) {
    auto P = [&](int i) { return (const float*)d_in[i]; };
    const float* x       = P(0);
    const float* bn_w    = P(1);  const float* bn_b   = P(2);
    const float* bn_mean = P(3);  const float* bn_var = P(4);
    const float* emb_W   = P(5);  const float* emb_b  = P(6);
    const float* conv1_W = P(7);  const float* conv1_b = P(8);
    const float* conv2_W = P(9);  const float* conv2_b = P(10);
    // stage bases: each stage dict = 16 arrays in insertion order
    const int S1 = 11, S2 = 27, S3 = 43;

    // --------------------- workspace carve (~102 MB) -----------------------
    const long NB_X = 32768L * 128;   // max h-wide activation elements
    const long NB_Z = 32768L * 256;   // max 2h-wide elements
    char* ws = (char*)d_ws;
    float*  X    = (float*)ws;
    float*  U    = X + NB_X;
    float*  Y    = U + NB_X;
    float*  Z    = Y + NB_X;
    __bf16* A_bf = (__bf16*)(Z + NB_Z);
    __bf16* W_bf = A_bf + NB_Z;

    auto EW = [](long n) { return dim3((unsigned)((n + 255) / 256)); };

    // ----------------------------- embed -----------------------------------
    {
        long n = 32768L * HDIM;
        k_bn_embed<<<EW(n), 256, 0, stream>>>(x, bn_w, bn_b, bn_mean, bn_var,
                                              emb_W, emb_b, X, n);
    }

    // --------------------------- stage runner ------------------------------
    auto stage = [&](int base, int h, int Lc) {
        const int  h2 = 2 * h;
        const long M  = (long)BATCH * Lc;
        const float* ln1_g = P(base + 0);  const float* ln1_b = P(base + 1);
        const float* logdt = P(base + 2);  const float* Arel  = P(base + 3);
        const float* Aim   = P(base + 4);  const float* Cre   = P(base + 5);
        const float* Cim   = P(base + 6);  const float* Dp    = P(base + 7);
        const float* Wout  = P(base + 8);  const float* bout  = P(base + 9);
        const float* ln2_g = P(base + 10); const float* ln2_b = P(base + 11);
        const float* W2    = P(base + 12); const float* b2    = P(base + 13);
        const float* W3    = P(base + 14); const float* b3    = P(base + 15);
        for (int j = 0; j < 2; ++j) {
            // --- S4 branch ---
            k_ln_rows<<<(unsigned)M, 32, 0, stream>>>(X, ln1_g + j * h, ln1_b + j * h, U, h);
            k_s4_fwd<<<BATCH * h, 32, 0, stream>>>(U, Y, logdt + j * h,
                Arel + (long)j * h * NSTATE, Aim + (long)j * h * NSTATE,
                Cre + (long)(j * 2 + 0) * h * NSTATE, Cim + (long)(j * 2 + 0) * h * NSTATE, Lc, h);
            k_s4_bwd<<<BATCH * h, 32, 0, stream>>>(U, Y, logdt + j * h,
                Arel + (long)j * h * NSTATE, Aim + (long)j * h * NSTATE,
                Cre + (long)(j * 2 + 1) * h * NSTATE, Cim + (long)(j * 2 + 1) * h * NSTATE,
                Dp + j * h, Lc, h);
            // z = gelu_out @ Wout^T + bout ; X += GLU(z)
            k_f32_to_bf16<<<EW(M * h), 256, 0, stream>>>(Y, A_bf, M * h);
            k_f32_to_bf16<<<EW((long)h2 * h), 256, 0, stream>>>(Wout + (long)j * h2 * h, W_bf, (long)h2 * h);
            k_wmma_gemm_bf16<<<dim3(h2 / 64, (unsigned)(M / 128)), 256, 0, stream>>>(
                A_bf, W_bf, bout + j * h2, Z, (int)M, h2, h);
            k_glu_residual<<<EW(M * h), 256, 0, stream>>>(Z, X, M * h, h);
            // --- MLP branch ---
            k_ln_rows<<<(unsigned)M, 32, 0, stream>>>(X, ln2_g + j * h, ln2_b + j * h, U, h);
            k_f32_to_bf16<<<EW(M * h), 256, 0, stream>>>(U, A_bf, M * h);
            k_f32_to_bf16<<<EW((long)h2 * h), 256, 0, stream>>>(W2 + (long)j * h2 * h, W_bf, (long)h2 * h);
            k_wmma_gemm_bf16<<<dim3(h2 / 64, (unsigned)(M / 128)), 256, 0, stream>>>(
                A_bf, W_bf, b2 + j * h2, Z, (int)M, h2, h);
            k_gelu_inplace<<<EW(M * h2), 256, 0, stream>>>(Z, M * h2);
            k_f32_to_bf16<<<EW(M * h2), 256, 0, stream>>>(Z, A_bf, M * h2);
            k_f32_to_bf16<<<EW((long)h * h2), 256, 0, stream>>>(W3 + (long)j * h * h2, W_bf, (long)h * h2);
            k_wmma_gemm_bf16<<<dim3(h / 64, (unsigned)(M / 128)), 256, 0, stream>>>(
                A_bf, W_bf, b3 + j * h, Y, (int)M, h, h2);
            k_add_inplace<<<EW(M * h), 256, 0, stream>>>(X, Y, M * h);
        }
    };

    auto down = [&](const float* cW, const float* cb, int Cin, int Cout, int Lin) {
        int Lo = Lin / RDOWN;
        long M = (long)BATCH * Lo;
        int K = Cin * RDOWN;
        k_conv_gather<<<EW(M * K), 256, 0, stream>>>(X, A_bf, Lo, Cin, M * K);
        k_f32_to_bf16<<<EW((long)Cout * K), 256, 0, stream>>>(cW, W_bf, (long)Cout * K);
        k_wmma_gemm_bf16<<<dim3(Cout / 64, (unsigned)(M / 128)), 256, 0, stream>>>(
            A_bf, W_bf, cb, Z, (int)M, Cout, K);
        k_copy<<<EW(M * Cout), 256, 0, stream>>>(Z, X, M * Cout);
    };

    stage(S1, 128, 4096);
    down(conv1_W, conv1_b, 128, 256, 4096);
    stage(S2, 256, 1024);
    down(conv2_W, conv2_b, 256, 512, 1024);
    stage(S3, 512, 256);

    k_upsample<<<EW((long)out_size), 256, 0, stream>>>(X, (float*)d_out, (long)out_size);
}
